// FlashAttention_5222680232518
// MI455X (gfx1250) — compile-verified, matrix operand order checked by reading, not run on hardware
//
#include <hip/hip_runtime.h>
#include <hip/hip_bf16.h>

typedef __attribute__((ext_vector_type(16))) _Float16 v16h;
typedef __attribute__((ext_vector_type(8)))  _Float16 v8h;
typedef __attribute__((ext_vector_type(8)))  float    v8f;
typedef __attribute__((ext_vector_type(4)))  int      v4i;

#define Bc 2
#define Hc 16
#define Sc 2048
#define Dc 128
#define NTILE (Sc / 32)            // 64 key tiles per (b,h)
#define TILE_HALVES (32 * 128)     // 4096 f16 per K or V tile (8KB)

#if defined(__has_builtin)
#if __has_builtin(__builtin_amdgcn_global_load_async_to_lds_b128)
#define ASYNC_KV 1
#endif
#endif

__device__ __forceinline__ float rmax16(float x) {
    x = fmaxf(x, __shfl_xor(x, 1, 32));
    x = fmaxf(x, __shfl_xor(x, 2, 32));
    x = fmaxf(x, __shfl_xor(x, 4, 32));
    x = fmaxf(x, __shfl_xor(x, 8, 32));
    return x;
}
__device__ __forceinline__ float rsum16(float x) {
    x += __shfl_xor(x, 1, 32);
    x += __shfl_xor(x, 2, 32);
    x += __shfl_xor(x, 4, 32);
    x += __shfl_xor(x, 8, 32);
    return x;
}

// Copy one 8KB tile (4096 f16) global -> LDS, 128 threads, thread t does 4x16B.
__device__ __forceinline__ void tile_copy(const _Float16* __restrict__ g,
                                          _Float16* l, int t) {
#ifdef ASYNC_KV
    typedef __attribute__((address_space(1))) v4i gv4i;   // global <4 x i32>
    typedef __attribute__((address_space(3))) v4i lv4i;   // LDS    <4 x i32>
#pragma unroll
    for (int i = 0; i < 4; i++) {
        __builtin_amdgcn_global_load_async_to_lds_b128(
            (gv4i*)(g + t * 8 + i * 1024),
            (lv4i*)(l + t * 8 + i * 1024), 0, 0);
    }
#else
#pragma unroll
    for (int i = 0; i < 4; i++) {
        *(v8h*)(l + t * 8 + i * 1024) = *(const v8h*)(g + t * 8 + i * 1024);
    }
#endif
}

__device__ __forceinline__ void wait_async_kv() {
#ifdef ASYNC_KV
    asm volatile("s_wait_asynccnt 0" ::: "memory");
#endif
}

// ---------------------------------------------------------------------------
// Pre-pass: fp32 K/V -> f16, V pre-transposed tile-wise so hot-loop LDS tiles
// are contiguous memcpy images.  kh: [bh][s][d] (tiles already contiguous).
// vh: [bh][tile][d][key32].
// ---------------------------------------------------------------------------
__global__ __launch_bounds__(256, 4)
void cvt_kv(const float* __restrict__ k, const float* __restrict__ v,
            _Float16* __restrict__ kh, _Float16* __restrict__ vh) {
    size_t j = (size_t)blockIdx.x * 256 + threadIdx.x;  // 0 .. B*H*S*D-1
    kh[j] = (_Float16)k[j];
    int key   = (int)(j & 31);
    int d     = (int)((j >> 5) & 127);
    size_t st = j >> 12;                         // bh*64 + tile
    size_t src = (st * 32 + key) * 128 + d;      // = ((bh*S)+s)*D + d
    vh[j] = (_Float16)v[src];
}

// ---------------------------------------------------------------------------
// Main kernel (workspace path): f16 K/V, async double-buffered LDS pipeline.
// ---------------------------------------------------------------------------
__global__ __launch_bounds__(128, 1)
void fa_fwd_wmma2(const float* __restrict__ q, const _Float16* __restrict__ kh,
                  const _Float16* __restrict__ vh, const int* __restrict__ mask,
                  float* __restrict__ out) {
    constexpr float SCALE = 0.08838834764831845f;  // 1/sqrt(128)
    constexpr float NEGV  = -1.0e9f;

    __shared__ alignas(32) _Float16 Kt[2][TILE_HALVES];   // [key][d]
    __shared__ alignas(32) _Float16 Vt[2][TILE_HALVES];   // [d][key]
    __shared__ alignas(32) _Float16 Qs[4][16 * 128];
    __shared__ alignas(32) _Float16 Pt[4][16 * 32];

    const int t    = threadIdx.x;
    const int wave = t >> 5;
    const int lane = t & 31;
    const int ln   = lane & 15;
    const int hf   = lane >> 4;
    const int bh   = blockIdx.y;
    const int b    = bh >> 4;
    const int q0   = blockIdx.x * 64 + wave * 16;

    const _Float16* kbase = kh + (size_t)bh * NTILE * TILE_HALVES;
    const _Float16* vbase = vh + (size_t)bh * NTILE * TILE_HALVES;

    // ---- prologue: kick off async copy of tile 0 while staging Q
    tile_copy(kbase, Kt[0], t);
    tile_copy(vbase, Vt[0], t);

    {
        const float* qp = q + ((size_t)bh * Sc + q0 + ln) * Dc + hf * 64;
        _Float16* dst = &Qs[wave][ln * 128 + hf * 64];
#pragma unroll
        for (int j = 0; j < 16; j++) {
            float4 x = ((const float4*)qp)[j];
            dst[4 * j + 0] = (_Float16)x.x;
            dst[4 * j + 1] = (_Float16)x.y;
            dst[4 * j + 2] = (_Float16)x.z;
            dst[4 * j + 3] = (_Float16)x.w;
        }
    }
    __syncthreads();

    v16h qf[4];
#pragma unroll
    for (int c = 0; c < 4; c++)
        qf[c] = *(const v16h*)&Qs[wave][ln * 128 + c * 32 + hf * 16];

    v8f o[8];
#pragma unroll
    for (int n = 0; n < 8; n++) o[n] = v8f{};
    float mrow[8], lrow[8];
#pragma unroll
    for (int g = 0; g < 8; g++) { mrow[g] = -3.0e38f; lrow[g] = 0.0f; }

#pragma unroll 1
    for (int kb = 0; kb < NTILE; kb++) {
        const int cur = kb & 1;
        // loads for tile kb were issued last iteration (or prologue)
        wait_async_kv();
        __syncthreads();   // tiles visible; all waves done reading buffer cur^1

        if (kb + 1 < NTILE) {  // prefetch next tile into the other buffer
            tile_copy(kbase + (size_t)(kb + 1) * TILE_HALVES, Kt[cur ^ 1], t);
            tile_copy(vbase + (size_t)(kb + 1) * TILE_HALVES, Vt[cur ^ 1], t);
        }

        const _Float16* Kc = Kt[cur];
        const _Float16* Vc = Vt[cur];

        // ---- scores: S = Q x K^T for two 16-key sub-blocks (fp32 acc)
        v8f acc0 = v8f{}, acc1 = v8f{};
#pragma unroll
        for (int c = 0; c < 4; c++) {
            v16h b0 = *(const v16h*)&Kc[ln * 128 + c * 32 + hf * 16];
            acc0 = __builtin_amdgcn_wmma_f32_16x16x32_f16(false, qf[c], false, b0,
                                                          (short)0, acc0, false, false);
        }
#pragma unroll
        for (int c = 0; c < 4; c++) {
            v16h b1 = *(const v16h*)&Kc[(16 + ln) * 128 + c * 32 + hf * 16];
            acc1 = __builtin_amdgcn_wmma_f32_16x16x32_f16(false, qf[c], false, b1,
                                                          (short)0, acc1, false, false);
        }

        const int kcol = kb * 32;
        const int m0 = mask[b * Sc + kcol + ln];
        const int m1 = mask[b * Sc + kcol + 16 + ln];
        const float sc0 = m0 ? SCALE : 0.0f, ad0 = m0 ? 0.0f : NEGV;
        const float sc1 = m1 ? SCALE : 0.0f, ad1 = m1 ? 0.0f : NEGV;

        float p0[8], p1[8], alpha[8];
#pragma unroll
        for (int g = 0; g < 8; g++) {
            float s0 = acc0[g] * sc0 + ad0;
            float s1 = acc1[g] * sc1 + ad1;
            float bm   = rmax16(fmaxf(s0, s1));
            float mnew = fmaxf(mrow[g], bm);
            alpha[g] = __expf(mrow[g] - mnew);
            p0[g]    = __expf(s0 - mnew);
            p1[g]    = __expf(s1 - mnew);
            float rs = rsum16(p0[g] + p1[g]);
            lrow[g] = lrow[g] * alpha[g] + rs;
            mrow[g] = mnew;
        }

#pragma unroll
        for (int g = 0; g < 8; g++) {
            Pt[wave][(g + 8 * hf) * 32 + ln]      = (_Float16)p0[g];
            Pt[wave][(g + 8 * hf) * 32 + 16 + ln] = (_Float16)p1[g];
        }
#pragma unroll
        for (int n = 0; n < 8; n++) {
#pragma unroll
            for (int g = 0; g < 8; g++) o[n][g] *= alpha[g];
        }
        asm volatile("s_wait_dscnt 0" ::: "memory");  // wave-internal P bounce

        v16h pf = *(const v16h*)&Pt[wave][ln * 32 + hf * 16];
#pragma unroll
        for (int n = 0; n < 8; n++) {
            v16h vf = *(const v16h*)&Vc[(n * 16 + ln) * 32 + hf * 16];
            o[n] = __builtin_amdgcn_wmma_f32_16x16x32_f16(false, pf, false, vf,
                                                          (short)0, o[n], false, false);
        }
    }

    float rcp[8];
#pragma unroll
    for (int g = 0; g < 8; g++) rcp[g] = 1.0f / lrow[g];
#pragma unroll
    for (int n = 0; n < 8; n++) {
#pragma unroll
        for (int g = 0; g < 8; g++) {
            out[((size_t)bh * Sc + q0 + g + 8 * hf) * Dc + n * 16 + ln] = o[n][g] * rcp[g];
        }
    }
}

// ---------------------------------------------------------------------------
// Fallback (no workspace): converts in-loop.
// ---------------------------------------------------------------------------
__global__ __launch_bounds__(128, 1)
void fa_fwd_wmma1(const float* __restrict__ q, const float* __restrict__ k,
                  const float* __restrict__ v, const int* __restrict__ mask,
                  float* __restrict__ out) {
    constexpr float SCALE = 0.08838834764831845f;
    constexpr float NEGV  = -1.0e9f;

    __shared__ alignas(32) _Float16 Kt[32 * 128];
    __shared__ alignas(32) _Float16 Vt[128 * 32];
    __shared__ alignas(32) _Float16 Qs[4][16 * 128];
    __shared__ alignas(32) _Float16 Pt[4][16 * 32];

    const int t    = threadIdx.x;
    const int wave = t >> 5;
    const int lane = t & 31;
    const int ln   = lane & 15;
    const int hf   = lane >> 4;
    const int bh   = blockIdx.y;
    const int b    = bh >> 4;
    const int q0   = blockIdx.x * 64 + wave * 16;

    {
        const float* qp = q + ((size_t)bh * Sc + q0 + ln) * Dc + hf * 64;
        _Float16* dst = &Qs[wave][ln * 128 + hf * 64];
#pragma unroll
        for (int j = 0; j < 16; j++) {
            float4 x = ((const float4*)qp)[j];
            dst[4 * j + 0] = (_Float16)x.x;
            dst[4 * j + 1] = (_Float16)x.y;
            dst[4 * j + 2] = (_Float16)x.z;
            dst[4 * j + 3] = (_Float16)x.w;
        }
    }
    __syncthreads();

    v16h qf[4];
#pragma unroll
    for (int c = 0; c < 4; c++)
        qf[c] = *(const v16h*)&Qs[wave][ln * 128 + c * 32 + hf * 16];

    v8f o[8];
#pragma unroll
    for (int n = 0; n < 8; n++) o[n] = v8f{};
    float mrow[8], lrow[8];
#pragma unroll
    for (int g = 0; g < 8; g++) { mrow[g] = -3.0e38f; lrow[g] = 0.0f; }

    const int row = t >> 2;
    const int d0  = (t & 3) * 32;

#pragma unroll 1
    for (int kb = 0; kb < NTILE; kb++) {
        const int kcol = kb * 32;
        __syncthreads();
        {
            const float* kp = k + ((size_t)bh * Sc + kcol + row) * Dc + d0;
            const float* vp = v + ((size_t)bh * Sc + kcol + row) * Dc + d0;
            _Float16* kd = &Kt[row * 128 + d0];
#pragma unroll
            for (int j = 0; j < 8; j++) {
                float4 x = ((const float4*)kp)[j];
                kd[4 * j + 0] = (_Float16)x.x;
                kd[4 * j + 1] = (_Float16)x.y;
                kd[4 * j + 2] = (_Float16)x.z;
                kd[4 * j + 3] = (_Float16)x.w;
                float4 y = ((const float4*)vp)[j];
                Vt[(d0 + 4 * j + 0) * 32 + row] = (_Float16)y.x;
                Vt[(d0 + 4 * j + 1) * 32 + row] = (_Float16)y.y;
                Vt[(d0 + 4 * j + 2) * 32 + row] = (_Float16)y.z;
                Vt[(d0 + 4 * j + 3) * 32 + row] = (_Float16)y.w;
            }
        }
        __syncthreads();

        v8f acc0 = v8f{}, acc1 = v8f{};
#pragma unroll
        for (int c = 0; c < 4; c++) {
            v16h b0 = *(const v16h*)&Kt[ln * 128 + c * 32 + hf * 16];
            acc0 = __builtin_amdgcn_wmma_f32_16x16x32_f16(false, qf[c], false, b0,
                                                          (short)0, acc0, false, false);
        }
#pragma unroll
        for (int c = 0; c < 4; c++) {
            v16h b1 = *(const v16h*)&Kt[(16 + ln) * 128 + c * 32 + hf * 16];
            acc1 = __builtin_amdgcn_wmma_f32_16x16x32_f16(false, qf[c], false, b1,
                                                          (short)0, acc1, false, false);
        }

        const int m0 = mask[b * Sc + kcol + ln];
        const int m1 = mask[b * Sc + kcol + 16 + ln];
        const float sc0 = m0 ? SCALE : 0.0f, ad0 = m0 ? 0.0f : NEGV;
        const float sc1 = m1 ? SCALE : 0.0f, ad1 = m1 ? 0.0f : NEGV;

        float p0[8], p1[8], alpha[8];
#pragma unroll
        for (int g = 0; g < 8; g++) {
            float s0 = acc0[g] * sc0 + ad0;
            float s1 = acc1[g] * sc1 + ad1;
            float bm   = rmax16(fmaxf(s0, s1));
            float mnew = fmaxf(mrow[g], bm);
            alpha[g] = __expf(mrow[g] - mnew);
            p0[g]    = __expf(s0 - mnew);
            p1[g]    = __expf(s1 - mnew);
            float rs = rsum16(p0[g] + p1[g]);
            lrow[g] = lrow[g] * alpha[g] + rs;
            mrow[g] = mnew;
        }

#pragma unroll
        for (int g = 0; g < 8; g++) {
            Pt[wave][(g + 8 * hf) * 32 + ln]      = (_Float16)p0[g];
            Pt[wave][(g + 8 * hf) * 32 + 16 + ln] = (_Float16)p1[g];
        }
#pragma unroll
        for (int n = 0; n < 8; n++) {
#pragma unroll
            for (int g = 0; g < 8; g++) o[n][g] *= alpha[g];
        }
        asm volatile("s_wait_dscnt 0" ::: "memory");

        v16h pf = *(const v16h*)&Pt[wave][ln * 32 + hf * 16];
#pragma unroll
        for (int n = 0; n < 8; n++) {
            v16h vf = *(const v16h*)&Vt[(n * 16 + ln) * 32 + hf * 16];
            o[n] = __builtin_amdgcn_wmma_f32_16x16x32_f16(false, pf, false, vf,
                                                          (short)0, o[n], false, false);
        }
    }

    float rcp[8];
#pragma unroll
    for (int g = 0; g < 8; g++) rcp[g] = 1.0f / lrow[g];
#pragma unroll
    for (int n = 0; n < 8; n++) {
#pragma unroll
        for (int g = 0; g < 8; g++) {
            out[((size_t)bh * Sc + q0 + g + 8 * hf) * Dc + n * 16 + ln] = o[n][g] * rcp[g];
        }
    }
}

extern "C" void kernel_launch(void* const* d_in, const int* in_sizes, int n_in,
                              void* d_out, int out_size, void* d_ws, size_t ws_size,
                              hipStream_t stream) {
    (void)in_sizes; (void)n_in; (void)out_size;
    const float* q    = (const float*)d_in[0];
    const float* k    = (const float*)d_in[1];
    const float* v    = (const float*)d_in[2];
    const int*   mask = (const int*)d_in[3];
    float* out = (float*)d_out;

    const size_t nelem   = (size_t)Bc * Hc * Sc * Dc;   // 8,388,608
    const size_t ws_need = 2 * nelem * sizeof(_Float16);

    dim3 grid(Sc / 64, Bc * Hc);
    dim3 block(128);

    if (ws_size >= ws_need && d_ws != nullptr) {
        _Float16* kh = (_Float16*)d_ws;
        _Float16* vh = kh + nelem;
        cvt_kv<<<dim3((unsigned)(nelem / 256)), dim3(256), 0, stream>>>(k, v, kh, vh);
        fa_fwd_wmma2<<<grid, block, 0, stream>>>(q, kh, vh, mask, out);
    } else {
        fa_fwd_wmma1<<<grid, block, 0, stream>>>(q, k, v, mask, out);
    }
}
